// Router_75368086110596
// MI455X (gfx1250) — matrix operand order repaired and checked by reading, not compiled, not run on hardware
//
#include <hip/hip_runtime.h>

typedef __attribute__((ext_vector_type(16))) __bf16 v16bf;
typedef __attribute__((ext_vector_type(8)))  __bf16 v8bf;
typedef __attribute__((ext_vector_type(8)))  float  v8f;
typedef __attribute__((ext_vector_type(4)))  float  v4f;

#define D_DIM 1024
#define E_DIM 64
#define WSTRIDE 1032          // 1024 + 8 halfs pad -> row stride 2064B, banks 4e%64
#define THREADS 256
#define TOK_PER_BLOCK 128     // 8 waves * 16 tokens

__device__ __forceinline__ v16bf lds_b_frag(const __bf16* p) {
  // 32 contiguous bytes from LDS -> two ds_load_b128
  v8bf lo = *(const v8bf*)p;
  v8bf hi = *(const v8bf*)(p + 8);
  v16bf r;
#pragma unroll
  for (int i = 0; i < 8; ++i) { r[i] = lo[i]; r[8 + i] = hi[i]; }
  return r;
}

__global__ __launch_bounds__(THREADS)
void router_top2_wmma(const float* __restrict__ X,
                      const float* __restrict__ W,
                      const float* __restrict__ bias,
                      float* __restrict__ out,
                      int n_tokens) {
  extern __shared__ __bf16 Wl[];   // [E_DIM][WSTRIDE] bf16, 129 KB

  const int tid = threadIdx.x;
  // Stage W (row-major [E][D] f32) into LDS as bf16, coalesced.
#pragma unroll 4
  for (int idx = tid; idx < E_DIM * D_DIM; idx += THREADS) {
    const int e = idx >> 10;
    const int d = idx & (D_DIM - 1);
    Wl[e * WSTRIDE + d] = (__bf16)W[idx];
  }
  __syncthreads();

  const int wave    = tid >> 5;
  const int lane    = tid & 31;
  const int half_id = lane >> 4;   // 0: lanes 0-15, 1: lanes 16-31
  const int ln      = lane & 15;
  const int token_base = blockIdx.x * TOK_PER_BLOCK + wave * 16;
  if (token_base >= n_tokens) return;

  const float* rowp = X + (size_t)(token_base + ln) * D_DIM;  // A row for M=ln
  const __bf16* b0p = &Wl[(0 * 16 + ln) * WSTRIDE];
  const __bf16* b1p = &Wl[(1 * 16 + ln) * WSTRIDE];
  const __bf16* b2p = &Wl[(2 * 16 + ln) * WSTRIDE];
  const __bf16* b3p = &Wl[(3 * 16 + ln) * WSTRIDE];

  v8f acc0 = {}, acc1 = {}, acc2 = {}, acc3 = {};

#pragma unroll 2
  for (int k0 = 0; k0 < D_DIM; k0 += 32) {
    // A 16x32 bf16 layout: lanes<16 hold K=k0+0..15,
    // lanes>=16 hold K=k0+16..31 (same k order as B).
    const int kA = k0 + half_id * 16;
    v4f a0 = *(const v4f*)(rowp + kA);
    v4f a1 = *(const v4f*)(rowp + kA + 4);
    v4f a2 = *(const v4f*)(rowp + kA + 8);
    v4f a3 = *(const v4f*)(rowp + kA + 12);
    v16bf A;
#pragma unroll
    for (int i = 0; i < 4; ++i) {
      A[i]      = (__bf16)a0[i];
      A[4 + i]  = (__bf16)a1[i];
      A[8 + i]  = (__bf16)a2[i];
      A[12 + i] = (__bf16)a3[i];
    }
    // B 32x16 bf16 layout: lanes<16 hold K=k0..k0+15, lanes>=16 hold K=k0+16..k0+31.
    const int kB = k0 + half_id * 16;
    v16bf B0 = lds_b_frag(b0p + kB);
    v16bf B1 = lds_b_frag(b1p + kB);
    v16bf B2 = lds_b_frag(b2p + kB);
    v16bf B3 = lds_b_frag(b3p + kB);

    acc0 = __builtin_amdgcn_wmma_f32_16x16x32_bf16(false, A, false, B0, (short)0, acc0, false, false);
    acc1 = __builtin_amdgcn_wmma_f32_16x16x32_bf16(false, A, false, B1, (short)0, acc1, false, false);
    acc2 = __builtin_amdgcn_wmma_f32_16x16x32_bf16(false, A, false, B2, (short)0, acc2, false, false);
    acc3 = __builtin_amdgcn_wmma_f32_16x16x32_bf16(false, A, false, B3, (short)0, acc3, false, false);
  }

  // Bias per expert column held by this lane.
  const float bb0 = bias[ln];
  const float bb1 = bias[16 + ln];
  const float bb2 = bias[32 + ln];
  const float bb3 = bias[48 + ln];

  // C layout: VGPR r holds token (r + half_id*8), expert = tile*16 + ln.
#pragma unroll
  for (int r = 0; r < 8; ++r) {
    float v0 = acc0[r] + bb0;
    float v1 = acc1[r] + bb1;
    float v2 = acc2[r] + bb2;
    float v3 = acc3[r] + bb3;

    // local top-2 of the 4 logits this lane owns for this token
    float hi1 = fmaxf(v0, v1), lo1 = fminf(v0, v1);
    float hi2 = fmaxf(v2, v3), lo2 = fminf(v2, v3);
    float m1 = fmaxf(hi1, hi2);
    float m2 = fmaxf(fminf(hi1, hi2), fmaxf(lo1, lo2));

    // merge (max, 2nd-max) across the 16 lanes sharing this token row;
    // xor masks 1,2,4,8 never cross the 16-lane half boundary in wave32.
#pragma unroll
    for (int s = 1; s <= 8; s <<= 1) {
      float o1 = __shfl_xor(m1, s, 32);
      float o2 = __shfl_xor(m2, s, 32);
      float n2 = fmaxf(fminf(m1, o1), fmaxf(m2, o2));
      m1 = fmaxf(m1, o1);
      m2 = n2;
    }

    // masked softmax: only the top-2 survive; Z = 1 + exp(m2 - m1)
    const float inv = 1.0f / (1.0f + __expf(m2 - m1));
    float* op = out + (size_t)(token_base + r + half_id * 8) * E_DIM + ln;
    op[0]  = (v0 >= m2) ? __expf(v0 - m1) * inv : 0.0f;
    op[16] = (v1 >= m2) ? __expf(v1 - m1) * inv : 0.0f;
    op[32] = (v2 >= m2) ? __expf(v2 - m1) * inv : 0.0f;
    op[48] = (v3 >= m2) ? __expf(v3 - m1) * inv : 0.0f;
  }
}

extern "C" void kernel_launch(void* const* d_in, const int* in_sizes, int n_in,
                              void* d_out, int out_size, void* d_ws, size_t ws_size,
                              hipStream_t stream) {
  const float* X    = (const float*)d_in[0];  // [B,S,D] f32
  const float* W    = (const float*)d_in[1];  // [E,D]   f32
  const float* bias = (const float*)d_in[2];  // [E]     f32
  float* out = (float*)d_out;                 // [B,S,E] f32

  const int n_tokens = in_sizes[0] / D_DIM;   // B*S = 32768
  const int blocks = (n_tokens + TOK_PER_BLOCK - 1) / TOK_PER_BLOCK;
  const size_t lds_bytes = (size_t)E_DIM * WSTRIDE * sizeof(__bf16);  // 132096 B

  hipLaunchKernelGGL(router_top2_wmma, dim3(blocks), dim3(THREADS), lds_bytes,
                     stream, X, W, bias, out, n_tokens);
}
